// Attention_37529424232765
// MI455X (gfx1250) — matrix operand order repaired and checked by reading, not compiled, not run on hardware
//
#include <hip/hip_runtime.h>
#include <hip/hip_bf16.h>

typedef __attribute__((ext_vector_type(16))) __bf16 v16bf;
typedef __attribute__((ext_vector_type(8)))  float  v8f;
typedef __attribute__((ext_vector_type(8)))  unsigned short ushort8;
typedef __attribute__((ext_vector_type(4)))  __bf16 bf4;
typedef __attribute__((ext_vector_type(4)))  float  f4;
typedef __attribute__((ext_vector_type(4)))  unsigned int uint4v;
typedef __attribute__((ext_vector_type(8)))  unsigned int uint8v;

union FragBF { v16bf v; ushort8 u[2]; };

__device__ __forceinline__ v8f vzero8() {
    v8f z = {0.f,0.f,0.f,0.f,0.f,0.f,0.f,0.f};
    return z;
}

// ---------------------------------------------------------------------------
// Tensor Data Mover: 2-D tile load Global -> LDS, bf16 (data_size = 2 bytes).
// Descriptor per CDNA5 ISA (D# groups 0/1).
// ---------------------------------------------------------------------------
__device__ __forceinline__ void tdm_load_2d_bf16(const void* lds_dst,
                                                 const void* global_src,
                                                 unsigned tensor_d0,   // elems
                                                 unsigned tensor_d1,   // rows
                                                 unsigned tile_d0,     // elems
                                                 unsigned tile_d1,     // rows
                                                 unsigned stride0)     // elems
{
    unsigned long long ga = (unsigned long long)(uintptr_t)global_src;
    unsigned lds_addr = (unsigned)(uintptr_t)lds_dst;   // low 32 bits = LDS offset

    uint4v g0;
    g0[0] = 1u;                                        // count=1, user mode
    g0[1] = lds_addr;                                  // lds_addr [63:32]
    g0[2] = (unsigned)(ga & 0xffffffffu);              // global_addr lo
    g0[3] = (unsigned)((ga >> 32) & 0x01ffffffu)       // global_addr [56:32]
          | (2u << 30);                                // type = 2 ("image")

    uint8v g1;
    g1[0] = (1u << 16);                                // data_size=1 -> 2 bytes
    g1[1] = (tensor_d0 & 0xffffu) << 16;               // tensor_dim0 [63:48]
    g1[2] = ((tensor_d0 >> 16) & 0xffffu)              // tensor_dim0 [79:64]
          | ((tensor_d1 & 0xffffu) << 16);             // tensor_dim1 [95:80]
    g1[3] = ((tensor_d1 >> 16) & 0xffffu)              // tensor_dim1 [111:96]
          | ((tile_d0 & 0xffffu) << 16);               // tile_dim0  [127:112]
    g1[4] = (tile_d1 & 0xffffu);                       // tile_dim1, tile_dim2=0
    g1[5] = stride0;                                   // dim0 stride lo32
    g1[6] = 0u;
    g1[7] = 0u;

    asm volatile("tensor_load_to_lds %0, %1" :: "s"(g0), "s"(g1) : "memory");
}

// A-fragment (16x32 bf16): lane = M row (lane&15), khalf = lane>>4.
__device__ __forceinline__ v16bf load_a_frag(const __bf16* rowbase, int lane) {
    int khalf = (lane >> 4) * 8;
    FragBF f;
    f.u[0] = *(const ushort8*)(rowbase + khalf);
    f.u[1] = *(const ushort8*)(rowbase + 16 + khalf);
    return f.v;
}

// B-fragment (32x16 bf16) from an LDS row holding K contiguously for one N.
__device__ __forceinline__ v16bf load_b_frag(const __bf16* rowbase, int lane) {
    int kb = (lane >> 4) * 8;
    FragBF f;
    f.u[0] = *(const ushort8*)(rowbase + kb);
    f.u[1] = *(const ushort8*)(rowbase + 16 + kb);
    return f.v;
}

// ---------------------------------------------------------------------------
// f32 -> bf16 convert, 4 elements per thread (n divisible by 1024)
// ---------------------------------------------------------------------------
__global__ void cvt_f32_bf16_kernel(const float* __restrict__ in,
                                    __bf16* __restrict__ out, int n) {
    int i = (blockIdx.x * blockDim.x + threadIdx.x) * 4;
    if (i + 3 < n) {
        f4 v = *(const f4*)(in + i);
        bf4 o;
        o.x = (__bf16)v.x; o.y = (__bf16)v.y;
        o.z = (__bf16)v.z; o.w = (__bf16)v.w;
        *(bf4*)(out + i) = o;
    }
}

// ---------------------------------------------------------------------------
// Tiled bf16 WMMA GEMM with double-buffered LDS:
//   A tile via TDM (wave 0, async, next buffer), B tile via register-held
//   global loads issued before compute, stored to the next buffer after.
// ---------------------------------------------------------------------------
#define BM 128
#define BN 128
#define BK 32

__global__ __launch_bounds__(256)
void gemm_bf16_wmma_kernel(const __bf16* __restrict__ A,
                           const __bf16* __restrict__ B,
                           const float*  __restrict__ bias,
                           float* __restrict__ outF,
                           __bf16* __restrict__ Qb,
                           __bf16* __restrict__ Kb,
                           __bf16* __restrict__ Vb,
                           int M, int N, int K, int mode)
{
    __shared__ __align__(16) __bf16 As[2][BM][BK];    // [m][k]  (TDM target)
    __shared__ __align__(16) __bf16 Bst[2][BN][BK];   // transposed: [n][k]

    const int tid  = threadIdx.x;
    const int lane = tid & 31;
    const int wave = tid >> 5;
    const int wm   = wave >> 2;
    const int wn   = wave & 3;
    const int m0   = blockIdx.y * BM;
    const int n0   = blockIdx.x * BN;

    const int kk0 = tid >> 4,          nc0 = (tid & 15) * 8;
    const int kk1 = (tid + 256) >> 4,  nc1 = (tid & 15) * 8;   // kk1 = kk0+16

    v8f acc[4][2];
#pragma unroll
    for (int mt = 0; mt < 4; ++mt)
#pragma unroll
        for (int nt = 0; nt < 2; ++nt) acc[mt][nt] = vzero8();

    // ---- prologue: stage tile k0 = 0 into buffer 0 ----
    if (wave == 0) {
        tdm_load_2d_bf16(&As[0][0][0], &A[(size_t)m0 * K],
                         (unsigned)K, (unsigned)M, BK, BM, (unsigned)K);
    }
    {
        ushort8 t0 = *(const ushort8*)&B[(size_t)kk0 * N + n0 + nc0];
        ushort8 t1 = *(const ushort8*)&B[(size_t)kk1 * N + n0 + nc1];
        const __bf16* p0 = (const __bf16*)&t0;
        const __bf16* p1 = (const __bf16*)&t1;
#pragma unroll
        for (int e = 0; e < 8; ++e) Bst[0][nc0 + e][kk0] = p0[e];
#pragma unroll
        for (int e = 0; e < 8; ++e) Bst[0][nc1 + e][kk1] = p1[e];
    }
    if (wave == 0) __builtin_amdgcn_s_wait_tensorcnt(0);
    __syncthreads();

    int buf = 0;
    for (int k0 = 0; k0 < K; k0 += BK) {
        const int cur = buf, nxt = buf ^ 1;
        const bool has_next = (k0 + BK) < K;

        // ---- issue next-tile staging before compute (hidden under WMMA) ----
        ushort8 t0, t1;
        if (has_next) {
            if (wave == 0) {
                tdm_load_2d_bf16(&As[nxt][0][0], &A[(size_t)m0 * K + k0 + BK],
                                 (unsigned)K, (unsigned)M, BK, BM, (unsigned)K);
            }
            t0 = *(const ushort8*)&B[(size_t)(k0 + BK + kk0) * N + n0 + nc0];
            t1 = *(const ushort8*)&B[(size_t)(k0 + BK + kk1) * N + n0 + nc1];
        }

        // ---- compute on current buffers ----
        v16bf bfr[2];
#pragma unroll
        for (int nt = 0; nt < 2; ++nt)
            bfr[nt] = load_b_frag(&Bst[cur][wn * 32 + nt * 16 + (lane & 15)][0], lane);

#pragma unroll
        for (int mt = 0; mt < 4; ++mt) {
            v16bf afr = load_a_frag(&As[cur][wm * 64 + mt * 16 + (lane & 15)][0], lane);
#pragma unroll
            for (int nt = 0; nt < 2; ++nt)
                acc[mt][nt] = __builtin_amdgcn_wmma_f32_16x16x32_bf16(
                    false, afr, false, bfr[nt], (short)0, acc[mt][nt], false, false);
        }

        // ---- finish staging into the next buffer, drain TDM, swap ----
        if (has_next) {
            const __bf16* p0 = (const __bf16*)&t0;
            const __bf16* p1 = (const __bf16*)&t1;
#pragma unroll
            for (int e = 0; e < 8; ++e) Bst[nxt][nc0 + e][kk0] = p0[e];
#pragma unroll
            for (int e = 0; e < 8; ++e) Bst[nxt][nc1 + e][kk1] = p1[e];
            if (wave == 0) __builtin_amdgcn_s_wait_tensorcnt(0);
        }
        __syncthreads();
        buf ^= 1;
    }

    // ---- epilogue ----
    const int nhalf = lane & 15;
    const int mhalf = (lane >> 4) * 8;
#pragma unroll
    for (int mt = 0; mt < 4; ++mt) {
#pragma unroll
        for (int nt = 0; nt < 2; ++nt) {
            int n = n0 + wn * 32 + nt * 16 + nhalf;
#pragma unroll
            for (int r = 0; r < 8; ++r) {
                int m = m0 + wm * 64 + mt * 16 + mhalf + r;
                float v = acc[mt][nt][r] + bias[n];
                if (mode == 0) {
                    outF[(size_t)m * N + n] = v;
                } else {
                    int which = n / 768;
                    int cc    = n - which * 768;
                    int h     = cc >> 6;
                    int d     = cc & 63;
                    int b     = m >> 10;
                    int t     = m & 1023;
                    size_t idx = (((size_t)(b * 12 + h)) * 1024 + t) * 64 + d;
                    __bf16 bv = (__bf16)v;
                    if (which == 0)      Qb[idx] = bv;
                    else if (which == 1) Kb[idx] = bv;
                    else                 Vb[idx] = bv;
                }
            }
        }
    }
}

// ---------------------------------------------------------------------------
// Attention: block per (b, h, 64-row tile), 4 waves. Streams ALL T columns
// (reference's multiplicative mask keeps masked cols = 1e-9 inside softmax).
// Double-buffered: K chunk via TDM, V chunk via register-held global loads.
// ---------------------------------------------------------------------------
__global__ __launch_bounds__(128)
void attn_wmma_kernel(const __bf16* __restrict__ Q,
                      const __bf16* __restrict__ Km,
                      const __bf16* __restrict__ V,
                      __bf16* __restrict__ Aout,
                      int Bn, int H, int T, int D)
{
    const int C = H * D;
    __shared__ __align__(16) __bf16 Kc[2][32][64];    // [j][d]  (TDM target)
    __shared__ __align__(16) __bf16 Vt[2][64][32];    // transposed: [d][j]
    __shared__ __align__(16) float  Sbuf[4][16][32];
    __shared__ __align__(16) __bf16 Pbuf[4][16][32];
    __shared__ float Mst[4][16], Lst[4][16], Ast[4][16];

    const int tid  = threadIdx.x;
    const int lane = tid & 31;
    const int w    = tid >> 5;
    const int t0   = blockIdx.x * 64;
    const int h    = blockIdx.y;
    const int b    = blockIdx.z;

    const size_t headoff = ((size_t)(b * H + h)) * (size_t)T * D;
    const __bf16* Qh = Q  + headoff;
    const __bf16* Kh = Km + headoff;
    const __bf16* Vh = V  + headoff;

    const int jj0 = tid >> 3,          dc0 = (tid & 7) * 8;
    const int jj1 = (tid + 128) >> 3,  dc1 = (tid & 7) * 8;    // jj1 = jj0+16

    // Q A-fragments for this wave's 16 rows
    v16bf aq[2];
    {
        int m = t0 + w * 16 + (lane & 15);
        int khalf = (lane >> 4) * 8;
#pragma unroll
        for (int ds = 0; ds < 2; ++ds) {
            const __bf16* qp = Qh + (size_t)m * D + ds * 32;
            FragBF f;
            f.u[0] = *(const ushort8*)(qp + khalf);
            f.u[1] = *(const ushort8*)(qp + 16 + khalf);
            aq[ds] = f.v;
        }
    }

    v8f o[4];
#pragma unroll
    for (int dt = 0; dt < 4; ++dt) o[dt] = vzero8();
    if (lane < 16) { Mst[w][lane] = -1e30f; Lst[w][lane] = 0.f; }

    // ---- prologue: stage chunk j0 = 0 into buffer 0 ----
    if (w == 0) {
        tdm_load_2d_bf16(&Kc[0][0][0], Kh,
                         (unsigned)D, (unsigned)T, (unsigned)D, 32, (unsigned)D);
    }
    {
        ushort8 tv0 = *(const ushort8*)&Vh[(size_t)jj0 * D + dc0];
        ushort8 tv1 = *(const ushort8*)&Vh[(size_t)jj1 * D + dc1];
        const __bf16* p0 = (const __bf16*)&tv0;
        const __bf16* p1 = (const __bf16*)&tv1;
#pragma unroll
        for (int e = 0; e < 8; ++e) Vt[0][dc0 + e][jj0] = p0[e];
#pragma unroll
        for (int e = 0; e < 8; ++e) Vt[0][dc1 + e][jj1] = p1[e];
    }
    if (w == 0) __builtin_amdgcn_s_wait_tensorcnt(0);
    __syncthreads();

    const int nhalf = lane & 15;
    const int mhalf = (lane >> 4) * 8;

    int buf = 0;
    for (int j0 = 0; j0 < T; j0 += 32) {
        const int cur = buf, nxt = buf ^ 1;
        const bool has_next = (j0 + 32) < T;

        // ---- issue next-chunk staging before compute ----
        ushort8 tv0, tv1;
        if (has_next) {
            if (w == 0) {
                tdm_load_2d_bf16(&Kc[nxt][0][0], Kh + (size_t)(j0 + 32) * D,
                                 (unsigned)D, (unsigned)T, (unsigned)D, 32,
                                 (unsigned)D);
            }
            tv0 = *(const ushort8*)&Vh[(size_t)(j0 + 32 + jj0) * D + dc0];
            tv1 = *(const ushort8*)&Vh[(size_t)(j0 + 32 + jj1) * D + dc1];
        }

        // ---- S = Q @ K^T ----
        v8f s[2] = { vzero8(), vzero8() };
#pragma unroll
        for (int nt = 0; nt < 2; ++nt) {
#pragma unroll
            for (int ds = 0; ds < 2; ++ds) {
                v16bf bf = load_b_frag(&Kc[cur][nt * 16 + nhalf][ds * 32], lane);
                s[nt] = __builtin_amdgcn_wmma_f32_16x16x32_bf16(
                    false, aq[ds], false, bf, (short)0, s[nt], false, false);
            }
        }

        // ---- multiplicative causal mask, stash S tile ----
#pragma unroll
        for (int nt = 0; nt < 2; ++nt) {
            int j = j0 + nt * 16 + nhalf;
#pragma unroll
            for (int r = 0; r < 8; ++r) {
                int i = t0 + w * 16 + mhalf + r;
                float sv = s[nt][r];
                if (j > i) sv = 1e-9f;
                Sbuf[w][mhalf + r][nt * 16 + nhalf] = sv;
            }
        }
        __syncthreads();

        // ---- online softmax update (one lane per row) ----
        if (lane < 16) {
            int r = lane;
            float mo = Mst[w][r];
            float cm = -1e30f;
#pragma unroll
            for (int n = 0; n < 32; ++n) cm = fmaxf(cm, Sbuf[w][r][n]);
            float mn = fmaxf(mo, cm);
            float al = __expf(mo - mn);
            float ssum = 0.f;
#pragma unroll
            for (int n = 0; n < 32; ++n) {
                float p = __expf(Sbuf[w][r][n] - mn);
                ssum += p;
                Pbuf[w][r][n] = (__bf16)p;
            }
            Mst[w][r] = mn;
            Lst[w][r] = Lst[w][r] * al + ssum;
            Ast[w][r] = al;
        }
        __syncthreads();

        // ---- rescale O, then O += P @ V ----
        float alr[8];
#pragma unroll
        for (int r = 0; r < 8; ++r) alr[r] = Ast[w][mhalf + r];
#pragma unroll
        for (int dt = 0; dt < 4; ++dt)
#pragma unroll
            for (int r = 0; r < 8; ++r) o[dt][r] *= alr[r];

        v16bf pf = load_a_frag(&Pbuf[w][lane & 15][0], lane);
#pragma unroll
        for (int dt = 0; dt < 4; ++dt) {
            v16bf bv = load_b_frag(&Vt[cur][dt * 16 + nhalf][0], lane);
            o[dt] = __builtin_amdgcn_wmma_f32_16x16x32_bf16(
                false, pf, false, bv, (short)0, o[dt], false, false);
        }

        // ---- finish staging next buffer, drain TDM, swap ----
        if (has_next) {
            const __bf16* p0 = (const __bf16*)&tv0;
            const __bf16* p1 = (const __bf16*)&tv1;
#pragma unroll
            for (int e = 0; e < 8; ++e) Vt[nxt][dc0 + e][jj0] = p0[e];
#pragma unroll
            for (int e = 0; e < 8; ++e) Vt[nxt][dc1 + e][jj1] = p1[e];
            if (w == 0) __builtin_amdgcn_s_wait_tensorcnt(0);
        }
        __syncthreads();
        buf ^= 1;
    }

    // ---- finalize: divide by l, write (d*H + h) interleaved layout ----
    float lr[8];
#pragma unroll
    for (int r = 0; r < 8; ++r) lr[r] = 1.0f / Lst[w][mhalf + r];
#pragma unroll
    for (int dt = 0; dt < 4; ++dt) {
        int d = dt * 16 + nhalf;
#pragma unroll
        for (int r = 0; r < 8; ++r) {
            int t = t0 + w * 16 + mhalf + r;
            Aout[((size_t)b * T + t) * C + (size_t)d * H + h] =
                (__bf16)(o[dt][r] * lr[r]);
        }
    }
}

// ---------------------------------------------------------------------------
// Host launcher
// ---------------------------------------------------------------------------
extern "C" void kernel_launch(void* const* d_in, const int* in_sizes, int n_in,
                              void* d_out, int out_size, void* d_ws, size_t ws_size,
                              hipStream_t stream)
{
    const float* x  = (const float*)d_in[0];
    const float* wa = (const float*)d_in[1];
    const float* ba = (const float*)d_in[2];
    const float* wp = (const float*)d_in[3];
    const float* bp = (const float*)d_in[4];
    float* out = (float*)d_out;

    const int Bn = 8, T = 1024, C = 768, H = 12, D = 64;
    const int M  = Bn * T;
    const int N3 = 3 * C;

    char* ws = (char*)d_ws;
    size_t off = 0;
    auto take = [&](size_t bytes) -> char* {
        char* p = ws + off;
        off += (bytes + 255) & ~(size_t)255;
        return p;
    };
    __bf16* xb  = (__bf16*)take((size_t)M * C * 2);
    __bf16* wab = (__bf16*)take((size_t)C * N3 * 2);
    __bf16* wpb = (__bf16*)take((size_t)C * C * 2);
    __bf16* Qb  = (__bf16*)take((size_t)M * C * 2);
    __bf16* Kb  = (__bf16*)take((size_t)M * C * 2);
    __bf16* Vb  = (__bf16*)take((size_t)M * C * 2);
    __bf16* Ab  = (__bf16*)take((size_t)M * C * 2);

    {
        int n = M * C;
        cvt_f32_bf16_kernel<<<dim3(n / 1024), 256, 0, stream>>>(x, xb, n);
    }
    {
        int n = C * N3;
        cvt_f32_bf16_kernel<<<dim3(n / 1024), 256, 0, stream>>>(wa, wab, n);
    }
    {
        int n = C * C;
        cvt_f32_bf16_kernel<<<dim3(n / 1024), 256, 0, stream>>>(wp, wpb, n);
    }

    gemm_bf16_wmma_kernel<<<dim3(N3 / BN, M / BM), 256, 0, stream>>>(
        xb, wab, ba, nullptr, Qb, Kb, Vb, M, N3, C, 1);

    attn_wmma_kernel<<<dim3(T / 64, H, Bn), 128, 0, stream>>>(
        Qb, Kb, Vb, Ab, Bn, H, T, D);

    gemm_bf16_wmma_kernel<<<dim3(C / BN, M / BM), 256, 0, stream>>>(
        Ab, wpb, bp, out, nullptr, nullptr, nullptr, M, C, C, 0);
}
